// CrossAttention_42253888258537
// MI455X (gfx1250) — matrix operand order repaired and checked by reading, not compiled, not run on hardware
//
#include <hip/hip_runtime.h>
#include <hip/hip_bf16.h>

// ---------------------------------------------------------------------------
// CrossAttention for MI455X (gfx1250, wave32, WMMA).
// All GEMMs via v_wmma_f32_16x16x32_f16 (f32 accumulate).
// Round-3 changes:
//  * GEMM: 8 waves/block share one weight colGroup (L0 reuse of W lines);
//    each wave now computes a 32x64 tile (8 wmma per 12 loads per k-step).
//  * Attention K/V staging uses global_load_async_to_lds_b128 (ASYNCcnt),
//    double-buffered: issue next chunk async, compute, s_wait_asynccnt, barrier.
// Workspace (~51MB).
// ---------------------------------------------------------------------------

typedef _Float16 h4  __attribute__((ext_vector_type(4)));
typedef _Float16 h8  __attribute__((ext_vector_type(8)));
typedef _Float16 v16h __attribute__((ext_vector_type(16)));
typedef float    v8f  __attribute__((ext_vector_type(8)));

#define DIM       512
#define NHEADS    8
#define HDIM      64
#define SEQ       4096
#define BATCH     2
#define ROWS      (BATCH * SEQ)       // 8192
#define LN10000   9.210340371976184f  // ln(ROPE_BASE)
#define SCALE     0.125f              // 64^-0.5

static __device__ __forceinline__ v16h ld16(const _Float16* p0, const _Float16* p1) {
    union { v16h v; h8 h[2]; } u;
    u.h[0] = *reinterpret_cast<const h8*>(p0);
    u.h[1] = *reinterpret_cast<const h8*>(p1);
    return u.v;
}

static __device__ __forceinline__ v8f wmma_f16(v16h a, v16h b, v8f c) {
    return __builtin_amdgcn_wmma_f32_16x16x32_f16(false, a, false, b,
                                                  (short)0, c, false, false);
}

// 16B global -> LDS async copy (CDNA5 TDM-adjacent path, tracked by ASYNCcnt).
// Low 32 bits of a generic shared-memory pointer are the LDS byte offset
// (ISA 10.2 aperture mapping).
static __device__ __forceinline__ void async_ld16(const _Float16* g, _Float16* l) {
#if defined(__gfx1250__)
    unsigned loff = (unsigned)(unsigned long long)l;
    unsigned long long ga = (unsigned long long)g;
    asm volatile("global_load_async_to_lds_b128 %0, %1, off"
                 :: "v"(loff), "v"(ga) : "memory");
#else
    *reinterpret_cast<h8*>(l) = *reinterpret_cast<const h8*>(g);
#endif
}

static __device__ __forceinline__ void wait_async() {
#if defined(__gfx1250__)
    asm volatile("s_wait_asynccnt 0x0" ::: "memory");
#endif
}

// ---------------------------------------------------------------------------
// fp32 -> fp16 convert (vectorized x4)
// ---------------------------------------------------------------------------
__global__ __launch_bounds__(256) void cvt_f32_f16(const float* __restrict__ s,
                                                   _Float16* __restrict__ d, int n4) {
    int i = blockIdx.x * 256 + threadIdx.x;
    if (i < n4) {
        float4 v = reinterpret_cast<const float4*>(s)[i];
        h4 o = { (_Float16)v.x, (_Float16)v.y, (_Float16)v.z, (_Float16)v.w };
        reinterpret_cast<h4*>(d)[i] = o;
    }
}

// ---------------------------------------------------------------------------
// RoPE table: tab[t*32+i] = (cos(t*invf[i]), sin(t*invf[i])), t<4096, i<32.
// ---------------------------------------------------------------------------
__global__ __launch_bounds__(256) void rope_tab_kernel(float2* __restrict__ tab) {
    int idx = blockIdx.x * 256 + threadIdx.x;  // 0..131071
    int t = idx >> 5, i = idx & 31;
    float inv = __expf(-(float)i * (LN10000 / 32.0f));
    float sn, cs;
    sincosf((float)t * inv, &sn, &cs);
    tab[idx] = make_float2(cs, sn);
}

// ---------------------------------------------------------------------------
// out = A(8192 x 512, f16) @ W^T(512x512, f16 [o][c]) + b
// One wave -> 32x64 tile.  256 blocks x 8 waves = 2048 waves.
// All waves of a block share colGroup -> identical W lines (L0 reuse).
// mode 0: bias + RoPE (table), f16 [B,H,seq,64]  (Q and K)
// mode 1: bias, f16 transposed [B,H,64,M]        (V)
// mode 2: bias, f32 row-major                    (final projection)
// ---------------------------------------------------------------------------
__global__ __launch_bounds__(256) void gemm512(const _Float16* __restrict__ A,
                                               const _Float16* __restrict__ W,
                                               const float* __restrict__ bias,
                                               const float2* __restrict__ rope,
                                               void* __restrict__ dst, int mode) {
    const int lane = threadIdx.x & 31;
    const int wave = threadIdx.x >> 5;
    const int ln = lane & 15, hi = lane >> 4;
    const int colGroup = blockIdx.x >> 5;                 // 0..7 (== head)
    const int rowPair  = (blockIdx.x & 31) * 8 + wave;    // 0..255 (32 rows each)

    const _Float16* arow0 = A + (size_t)(rowPair * 32 + ln) * DIM;
    const _Float16* arow1 = arow0 + (size_t)16 * DIM;

    v8f acc[2][4];
#pragma unroll
    for (int rt = 0; rt < 2; ++rt)
#pragma unroll
        for (int dt = 0; dt < 4; ++dt) acc[rt][dt] = (v8f){};

    for (int k = 0; k < DIM; k += 32) {
        v16h a0 = ld16(arow0 + k + 8 * hi, arow0 + k + 8 * hi + 16);
        v16h a1 = ld16(arow1 + k + 8 * hi, arow1 + k + 8 * hi + 16);
#pragma unroll
        for (int dt = 0; dt < 4; ++dt) {
            const _Float16* wrow =
                W + (size_t)(colGroup * 64 + dt * 16 + ln) * DIM + k + 8 * hi;
            v16h b = ld16(wrow, wrow + 16);
            acc[0][dt] = wmma_f16(a0, b, acc[0][dt]);
            acc[1][dt] = wmma_f16(a1, b, acc[1][dt]);
        }
    }

    const int h = colGroup;

#pragma unroll
    for (int rt = 0; rt < 2; ++rt) {
        const int rowTile = rowPair * 2 + rt;  // 16-row tile index
        const int b_idx = rowTile >> 8;
#pragma unroll
        for (int dt = 0; dt < 4; ++dt) {
            const int o = colGroup * 64 + dt * 16 + ln;  // output channel
            const float bv = bias[o];
            if (mode == 0) {  // RoPE + head-major f16 (Q / K)
                const int d = o & 63;
                _Float16* q = (_Float16*)dst;
#pragma unroll
                for (int r = 0; r < 8; ++r) {
                    const int n = rowTile * 16 + r + 8 * hi;
                    const int t = n & (SEQ - 1);
                    float val = acc[rt][dt][r] + bv;
                    float pv = __shfl_xor(val, 1);  // partner d^1 (w/ its bias)
                    float2 cssn = rope[t * 32 + (d & 31)];
                    float rot = (d & 1) ? pv : -pv;  // interleaved rotate_half
                    q[(((size_t)b_idx * NHEADS + h) * SEQ + t) * HDIM + d] =
                        (_Float16)(val * cssn.x + rot * cssn.y);
                }
            } else if (mode == 1) {  // transposed f16 (V^T): lane owns d
                const int d = dt * 16 + ln;
                h8 pk;
#pragma unroll
                for (int r = 0; r < 8; ++r)
                    pk[r] = (_Float16)(acc[rt][dt][r] + bv);
                const int m = (rowTile * 16 + 8 * hi) & (SEQ - 1);
                _Float16* vt = (_Float16*)dst;
                *reinterpret_cast<h8*>(
                    vt + (((size_t)b_idx * NHEADS + h) * HDIM + d) * SEQ + m) = pk;
            } else {  // f32 row-major (final output)
                float* outp = (float*)dst;
#pragma unroll
                for (int r = 0; r < 8; ++r) {
                    const int n = rowTile * 16 + r + 8 * hi;
                    outp[(size_t)n * DIM + o] = acc[rt][dt][r] + bv;
                }
            }
        }
    }
}

// ---------------------------------------------------------------------------
// Flash attention.  256 blocks x 8 waves; all waves of a block share (b,h);
// K/V chunks double-buffered in LDS via global_load_async_to_lds_b128.
// Per wave: 32 query rows; per 32-col chunk: 8 wmma (S) + softmax + 8 wmma (PV).
// ---------------------------------------------------------------------------
__global__ __launch_bounds__(256) void attn64(const _Float16* __restrict__ Q,
                                              const _Float16* __restrict__ K,
                                              const _Float16* __restrict__ Vt,
                                              _Float16* __restrict__ attnOut) {
    __shared__ __align__(16) _Float16 Kbuf[2][32][72];  // 32 m x 64 d (+pad)
    __shared__ __align__(16) _Float16 Vbuf[2][64][40];  // 64 d x 32 m (+pad)
    __shared__ __align__(16) _Float16 Pbuf[8][32][32];  // per-wave P tile

    const int tid = threadIdx.x;
    const int lane = tid & 31;
    const int wave = tid >> 5;
    const int ln = lane & 15, hi = lane >> 4;

    const int bh = blockIdx.x >> 4;        // 0..15  (b*8+h)
    const int grp = blockIdx.x & 15;       // query-block group
    const int nb = grp * 8 + wave;         // 0..127 : this wave's 32-row block
    const int b = bh >> 3, h = bh & 7;

    const _Float16* Qb = Q + ((size_t)bh * SEQ + nb * 32) * HDIM;
    const _Float16* Kb = K + (size_t)bh * SEQ * HDIM;
    const _Float16* Vb = Vt + (size_t)bh * HDIM * SEQ;

    // cooperative-load coordinates (per thread: one 16B piece of K and of V)
    const int krow = tid >> 3, kcol = (tid & 7) * 8;   // K: 32 x 64
    const int vrow = tid >> 2, vcol = (tid & 3) * 8;   // V: 64 x 32

    // Q A-fragments: 2 n-tiles x 2 k-chunks (d 0..31 / 32..63)
    v16h qa[2][2];
#pragma unroll
    for (int t = 0; t < 2; ++t)
#pragma unroll
        for (int kc = 0; kc < 2; ++kc) {
            const _Float16* p = Qb + (size_t)(t * 16 + ln) * HDIM + kc * 32 + 8 * hi;
            qa[t][kc] = ld16(p, p + 16);
        }

    v8f o_acc[2][4];
    float rmax[2][8], rsum[2][8];
#pragma unroll
    for (int t = 0; t < 2; ++t) {
#pragma unroll
        for (int dt = 0; dt < 4; ++dt) o_acc[t][dt] = (v8f){};
#pragma unroll
        for (int r = 0; r < 8; ++r) { rmax[t][r] = -3.0e38f; rsum[t][r] = 0.0f; }
    }

    // async-preload chunk 0 into buffer 0
    async_ld16(Kb + (size_t)krow * HDIM + kcol, &Kbuf[0][krow][kcol]);
    async_ld16(Vb + (size_t)vrow * SEQ + vcol, &Vbuf[0][vrow][vcol]);
    wait_async();
    __syncthreads();

    int buf = 0;
    for (int mc = 0; mc < SEQ; mc += 32) {
        // kick off next chunk's async copies into the other buffer
        // (that buffer was last READ before the previous barrier -> safe)
        const int mcn = (mc + 32) & (SEQ - 1);
        async_ld16(Kb + (size_t)(mcn + krow) * HDIM + kcol,
                   &Kbuf[buf ^ 1][krow][kcol]);
        async_ld16(Vb + (size_t)vrow * SEQ + mcn + vcol,
                   &Vbuf[buf ^ 1][vrow][vcol]);

        // --- S = scale * Q K^T  (32x32), K from LDS ---
        v16h kf[2][2];
#pragma unroll
        for (int j = 0; j < 2; ++j)
#pragma unroll
            for (int kc = 0; kc < 2; ++kc) {
                const _Float16* p = &Kbuf[buf][j * 16 + ln][kc * 32 + 8 * hi];
                kf[j][kc] = ld16(p, p + 16);
            }
        v8f s[2][2];
#pragma unroll
        for (int t = 0; t < 2; ++t)
#pragma unroll
            for (int j = 0; j < 2; ++j) {
                v8f z = (v8f){};
                z = wmma_f16(qa[t][0], kf[j][0], z);
                z = wmma_f16(qa[t][1], kf[j][1], z);
#pragma unroll
                for (int r = 0; r < 8; ++r) z[r] *= SCALE;
                s[t][j] = z;
            }

        // --- online softmax ---
#pragma unroll
        for (int t = 0; t < 2; ++t) {
            float corr[8];
#pragma unroll
            for (int r = 0; r < 8; ++r) {
                float mloc = fmaxf(s[t][0][r], s[t][1][r]);
                mloc = fmaxf(mloc, __shfl_xor(mloc, 1));
                mloc = fmaxf(mloc, __shfl_xor(mloc, 2));
                mloc = fmaxf(mloc, __shfl_xor(mloc, 4));
                mloc = fmaxf(mloc, __shfl_xor(mloc, 8));
                const float nm = fmaxf(rmax[t][r], mloc);
                corr[r] = __expf(rmax[t][r] - nm);
                rmax[t][r] = nm;
                const float p0 = __expf(s[t][0][r] - nm);
                const float p1 = __expf(s[t][1][r] - nm);
                s[t][0][r] = p0;
                s[t][1][r] = p1;
                float lsum = p0 + p1;
                lsum += __shfl_xor(lsum, 1);
                lsum += __shfl_xor(lsum, 2);
                lsum += __shfl_xor(lsum, 4);
                lsum += __shfl_xor(lsum, 8);
                rsum[t][r] = rsum[t][r] * corr[r] + lsum;
            }
#pragma unroll
            for (int dt = 0; dt < 4; ++dt)
#pragma unroll
                for (int r = 0; r < 8; ++r) o_acc[t][dt][r] *= corr[r];
#pragma unroll
            for (int j = 0; j < 2; ++j)
#pragma unroll
                for (int r = 0; r < 8; ++r)
                    Pbuf[wave][t * 16 + r + 8 * hi][j * 16 + ln] =
                        (_Float16)s[t][j][r];
        }
#if defined(__gfx1250__)
        asm volatile("s_wait_dscnt 0" ::: "memory");
#endif
        // --- O += P V  (P from per-wave LDS, V from shared LDS) ---
        v16h pa[2];
#pragma unroll
        for (int t = 0; t < 2; ++t) {
            const _Float16* p = &Pbuf[wave][t * 16 + ln][8 * hi];
            pa[t] = ld16(p, p + 16);
        }
        v16h vf[4];
#pragma unroll
        for (int dt = 0; dt < 4; ++dt) {
            const _Float16* p = &Vbuf[buf][dt * 16 + ln][8 * hi];
            vf[dt] = ld16(p, p + 16);
        }
#pragma unroll
        for (int t = 0; t < 2; ++t)
#pragma unroll
            for (int dt = 0; dt < 4; ++dt)
                o_acc[t][dt] = wmma_f16(pa[t], vf[dt], o_acc[t][dt]);

        // next buffer must be fully landed before anyone reads it
        wait_async();
        __syncthreads();
        buf ^= 1;
    }

    // --- normalize + store f16 attn output [B,N,512] ---
#pragma unroll
    for (int t = 0; t < 2; ++t)
#pragma unroll
        for (int r = 0; r < 8; ++r) {
            const float inv = 1.0f / rsum[t][r];
            const int n = nb * 32 + t * 16 + r + 8 * hi;
#pragma unroll
            for (int dt = 0; dt < 4; ++dt)
                attnOut[((size_t)b * SEQ + n) * DIM + h * 64 + dt * 16 + ln] =
                    (_Float16)(o_acc[t][dt][r] * inv);
        }
}

// ---------------------------------------------------------------------------
extern "C" void kernel_launch(void* const* d_in, const int* in_sizes, int n_in,
                              void* d_out, int out_size, void* d_ws, size_t ws_size,
                              hipStream_t stream) {
    (void)in_sizes; (void)n_in; (void)out_size; (void)ws_size;
    const float* x   = (const float*)d_in[0];
    const float* ctx = (const float*)d_in[1];
    const float* Wq  = (const float*)d_in[2];
    const float* bq  = (const float*)d_in[3];
    const float* Wk  = (const float*)d_in[4];
    const float* bk  = (const float*)d_in[5];
    const float* Wv  = (const float*)d_in[6];
    const float* bv  = (const float*)d_in[7];
    const float* Wo  = (const float*)d_in[8];
    const float* bo  = (const float*)d_in[9];

    char* ws = (char*)d_ws;
    const size_t ACT = (size_t)ROWS * DIM;  // 4194304 elements
    const size_t WSZ = (size_t)DIM * DIM;   // 262144 elements
    size_t off = 0;
    _Float16* xh  = (_Float16*)(ws + off); off += ACT * 2;
    _Float16* ch  = (_Float16*)(ws + off); off += ACT * 2;
    _Float16* wqh = (_Float16*)(ws + off); off += WSZ * 2;
    _Float16* wkh = (_Float16*)(ws + off); off += WSZ * 2;
    _Float16* wvh = (_Float16*)(ws + off); off += WSZ * 2;
    _Float16* woh = (_Float16*)(ws + off); off += WSZ * 2;
    _Float16* qb  = (_Float16*)(ws + off); off += ACT * 2;
    _Float16* kb  = (_Float16*)(ws + off); off += ACT * 2;
    _Float16* vt  = (_Float16*)(ws + off); off += ACT * 2;
    _Float16* ab  = (_Float16*)(ws + off); off += ACT * 2;
    float2*   rt  = (float2*)(ws + off);   off += (size_t)SEQ * 32 * 8;  // ~51 MB

    // fp32 -> fp16 + RoPE table
    cvt_f32_f16<<<(int)(ACT / 1024), 256, 0, stream>>>(x,   xh, (int)(ACT / 4));
    cvt_f32_f16<<<(int)(ACT / 1024), 256, 0, stream>>>(ctx, ch, (int)(ACT / 4));
    cvt_f32_f16<<<(int)(WSZ / 1024), 256, 0, stream>>>(Wq, wqh, (int)(WSZ / 4));
    cvt_f32_f16<<<(int)(WSZ / 1024), 256, 0, stream>>>(Wk, wkh, (int)(WSZ / 4));
    cvt_f32_f16<<<(int)(WSZ / 1024), 256, 0, stream>>>(Wv, wvh, (int)(WSZ / 4));
    cvt_f32_f16<<<(int)(WSZ / 1024), 256, 0, stream>>>(Wo, woh, (int)(WSZ / 4));
    rope_tab_kernel<<<512, 256, 0, stream>>>(rt);

    // projections (+RoPE for Q,K; transpose for V)
    gemm512<<<256, 256, 0, stream>>>(xh, wqh, bq, rt, qb, 0);
    gemm512<<<256, 256, 0, stream>>>(ch, wkh, bk, rt, kb, 0);
    gemm512<<<256, 256, 0, stream>>>(ch, wvh, bv, rt, vt, 1);

    // flash attention
    attn64<<<256, 256, 0, stream>>>(qb, kb, vt, ab);

    // output projection -> f32
    gemm512<<<256, 256, 0, stream>>>(ab, woh, bo, rt, d_out, 2);
}
